// multi_head_attention_45913200394553
// MI455X (gfx1250) — matrix operand order repaired and checked by reading, not compiled, not run on hardware
//
#include <hip/hip_runtime.h>

#define B_  4
#define D_  1024
#define L_  1024
#define H_  16
#define DH_ 64

typedef __attribute__((ext_vector_type(16))) __bf16         v16bf;
typedef __attribute__((ext_vector_type(16))) unsigned short v16us;
typedef __attribute__((ext_vector_type(8)))  float          v8f;
typedef __attribute__((ext_vector_type(4)))  unsigned int   v4u;

union BF16x16 { v16us u; v16bf b; v4u q[2]; };

__device__ __forceinline__ unsigned short f2bf(float x) {
  unsigned u = __float_as_uint(x);
  unsigned r = u + 0x7FFFu + ((u >> 16) & 1u);   // round-to-nearest-even
  return (unsigned short)(r >> 16);
}
__device__ __forceinline__ float bf2f(unsigned short h) {
  return __uint_as_float(((unsigned)h) << 16);
}
__device__ __forceinline__ void splitStore(float x, BF16x16 &hi, BF16x16 &lo, int e) {
  unsigned short h = f2bf(x);
  hi.u[e] = h;
  lo.u[e] = f2bf(x - bf2f(h));
}
// 16 contiguous bf16 -> operand register (2 x b128)
__device__ __forceinline__ void load16(const unsigned short* p, BF16x16 &d) {
  const v4u* q = (const v4u*)p;
  d.q[0] = q[0]; d.q[1] = q[1];
}
// A-style natural row: elements 0..7 at p[0..7], elements 8..15 at p[16..23]
__device__ __forceinline__ void load8_8(const unsigned short* p, BF16x16 &d) {
  const v4u* q = (const v4u*)p;
  d.q[0] = q[0]; d.q[1] = q[2];
}

__device__ __forceinline__ v8f wmma_bf16(v16bf a, v16bf b, v8f c) {
  return __builtin_amdgcn_wmma_f32_16x16x32_bf16(false, a, false, b, (short)0, c, false, false);
}

// ---------------------------------------------------------------------------
// Pack kernels: fp32 -> bf16 hi/lo planes in WMMA-operand order.
// Operand tile = 512 bf16 (32 lanes x 16 elements), one K-step of 32.
// ---------------------------------------------------------------------------

// Activations Q,K,V -> B-operand planes.
// tid = ((((mat*4+b)*32+kt)*64+nt)*32+lane)*16+e ; k=kt*32+(lane>>4)*16+e ; n=nt*16+(lane&15)
__global__ void __launch_bounds__(256) pack_act_kernel(
    const float* __restrict__ Q, const float* __restrict__ K, const float* __restrict__ V,
    unsigned short* __restrict__ hi, unsigned short* __restrict__ lo)
{
  const unsigned t = blockIdx.x * 256 + threadIdx.x;   // < 12582912
  const int e    = t & 15;
  const int lane = (t >> 4) & 31;
  const int nt   = (t >> 9) & 63;
  const int kt   = (t >> 15) & 31;
  const int b    = (t >> 20) & 3;
  const int mat  = t >> 22;
  const float* X = (mat == 0) ? Q : (mat == 1) ? K : V;
  const int k = kt * 32 + ((lane >> 4) << 4) + e;
  const int n = nt * 16 + (lane & 15);
  const float x = X[((size_t)b * D_ + k) * L_ + n];
  const unsigned short h = f2bf(x);
  hi[t] = h;
  lo[t] = f2bf(x - bf2f(h));
}

// Weights WQ,WK,WV -> A-operand planes.
// tid = ((((mat*16+h)*32+kt)*4+mt)*32+lane)*16+e ; m=mt*16+(lane&15) ;
// k = kt*32 + (lane>>4)*16 + e
__global__ void __launch_bounds__(256) pack_w_kernel(
    const float* __restrict__ WQ, const float* __restrict__ WK, const float* __restrict__ WV,
    unsigned short* __restrict__ hi, unsigned short* __restrict__ lo)
{
  const unsigned t = blockIdx.x * 256 + threadIdx.x;   // < 3145728
  const int e    = t & 15;
  const int lane = (t >> 4) & 31;
  const int mt   = (t >> 9) & 3;
  const int kt   = (t >> 11) & 31;
  const int h    = (t >> 16) & 15;
  const int mat  = t >> 20;
  const float* W = (mat == 0) ? WQ : (mat == 1) ? WK : WV;
  const int m = mt * 16 + (lane & 15);
  const int k = kt * 32 + (lane >> 4) * 16 + e;
  const float x = W[((size_t)h * DH_ + m) * D_ + k];
  const unsigned short hv = f2bf(x);
  hi[t] = hv;
  lo[t] = f2bf(x - bf2f(hv));
}

// Wo -> A-operand planes.  tid = ((mbig*32+kt)*32+lane)*16+e ; m=mbig*16+(lane&15)
__global__ void __launch_bounds__(256) pack_wo_kernel(
    const float* __restrict__ Wo,
    unsigned short* __restrict__ hi, unsigned short* __restrict__ lo)
{
  const unsigned t = blockIdx.x * 256 + threadIdx.x;   // < 1048576
  const int e    = t & 15;
  const int lane = (t >> 4) & 31;
  const int kt   = (t >> 9) & 31;
  const int mbig = t >> 14;
  const int m = mbig * 16 + (lane & 15);
  const int k = kt * 32 + (lane >> 4) * 16 + e;
  const float x = Wo[(size_t)m * D_ + k];
  const unsigned short hv = f2bf(x);
  hi[t] = hv;
  lo[t] = f2bf(x - bf2f(hv));
}

// ---------------------------------------------------------------------------
// Kernel 1: QKV projections from packed operands.
// Outputs bf16 hi/lo planes: Q,K transposed (BH,L,DH); V as (BH,DH,L).
// One wave: 64(M) x 16(N) tile, K-loop over D in steps of 32.
// ---------------------------------------------------------------------------
__global__ void __launch_bounds__(256) proj_qkv_kernel(
    const unsigned short* __restrict__ Xp_hi, const unsigned short* __restrict__ Xp_lo,
    const unsigned short* __restrict__ Wp_hi, const unsigned short* __restrict__ Wp_lo,
    unsigned short* __restrict__ Qh, unsigned short* __restrict__ Ql,
    unsigned short* __restrict__ Kh, unsigned short* __restrict__ Kl,
    unsigned short* __restrict__ Vh, unsigned short* __restrict__ Vl)
{
  const int wid  = blockIdx.x * 8 + (threadIdx.x >> 5);
  const int lane = threadIdx.x & 31;
  const int lm = lane & 15;
  const int hh = lane >> 4;

  const int ntile = wid & 63;
  const int mat   = (wid >> 6) % 3;
  const int bh    = wid / 192;
  const int b = bh >> 4, h = bh & 15;

  const int n_abs = ntile * 16 + lm;

  v8f acc[4];
#pragma unroll
  for (int mt = 0; mt < 4; ++mt)
#pragma unroll
    for (int r = 0; r < 8; ++r) acc[mt][r] = 0.f;

  for (int kt = 0; kt < 32; ++kt) {
    const size_t boff = (((((size_t)mat * 4 + b) * 32 + kt) * 64 + ntile) * 32 + lane) * 16;
    BF16x16 bhi, blo;
    load16(Xp_hi + boff, bhi);
    load16(Xp_lo + boff, blo);
#pragma unroll
    for (int mt = 0; mt < 4; ++mt) {
      const size_t aoff = (((((size_t)mat * 16 + h) * 32 + kt) * 4 + mt) * 32 + lane) * 16;
      BF16x16 ahi, alo;
      load16(Wp_hi + aoff, ahi);
      load16(Wp_lo + aoff, alo);
      acc[mt] = wmma_bf16(ahi.b, bhi.b, acc[mt]);
      acc[mt] = wmma_bf16(ahi.b, blo.b, acc[mt]);
      acc[mt] = wmma_bf16(alo.b, bhi.b, acc[mt]);
    }
  }

#pragma unroll
  for (int mt = 0; mt < 4; ++mt) {
#pragma unroll
    for (int r = 0; r < 8; ++r) {
      const int m = mt * 16 + r + 8 * hh;
      const float v = acc[mt][r];
      const unsigned short hv = f2bf(v);
      const unsigned short lv = f2bf(v - bf2f(hv));
      if (mat == 2) {
        const size_t o = ((size_t)bh * DH_ + m) * L_ + n_abs;
        Vh[o] = hv; Vl[o] = lv;
      } else if (mat == 0) {
        const size_t o = ((size_t)bh * L_ + n_abs) * DH_ + m;
        Qh[o] = hv; Ql[o] = lv;
      } else {
        const size_t o = ((size_t)bh * L_ + n_abs) * DH_ + m;
        Kh[o] = hv; Kl[o] = lv;
      }
    }
  }
}

// ---------------------------------------------------------------------------
// Kernel 2: flash attention (swapped Q/K form == reference row-axis softmax).
// S'[l][m] = sum_d Qd[d,l]*Kd[d,m]/8, causal l<=m, online softmax over l,
// O[d][m] += Vd[d,l]*P[l][m].  Output stored transposed as bf16 catT (B,L,D).
// ---------------------------------------------------------------------------
__global__ void __launch_bounds__(256) attn_kernel(
    const unsigned short* __restrict__ Qh, const unsigned short* __restrict__ Ql,
    const unsigned short* __restrict__ Kh, const unsigned short* __restrict__ Kl,
    const unsigned short* __restrict__ Vh, const unsigned short* __restrict__ Vl,
    unsigned short* __restrict__ Ch, unsigned short* __restrict__ Cl)
{
  const int wid  = blockIdx.x * 8 + (threadIdx.x >> 5);
  const int lane = threadIdx.x & 31;
  const int lm = lane & 15, hh = lane >> 4;
  const int mtile = wid & 63;
  const int bh    = wid >> 6;
  const int m0 = mtile * 16;
  const int m_abs = m0 + lm;

  v8f O[4];
#pragma unroll
  for (int dt = 0; dt < 4; ++dt)
#pragma unroll
    for (int r = 0; r < 8; ++r) O[dt][r] = 0.f;

  float run_max = -1e30f, run_sum = 0.f;

  const int nchunk = (m0 + 47) / 32;
  for (int c = 0; c < nchunk; ++c) {
    const int lb = c * 32;

    v8f Sa, Sb;
#pragma unroll
    for (int r = 0; r < 8; ++r) { Sa[r] = 0.f; Sb[r] = 0.f; }

#pragma unroll
    for (int dd = 0; dd < 2; ++dd) {
      const int d0 = dd * 32;
      // B operand: KdT row m_abs, 16 contiguous d
      const size_t krow = ((size_t)bh * L_ + m_abs) * DH_ + d0 + hh * 16;
      BF16x16 bhi, blo;
      load16(Kh + krow, bhi);
      load16(Kl + krow, blo);
#pragma unroll
      for (int t = 0; t < 2; ++t) {
        // A operand: QdT row (lb + t*16 + lm), 16 contiguous d
        const size_t qrow = ((size_t)bh * L_ + lb + t * 16 + lm) * DH_ + d0 + hh * 16;
        BF16x16 ahi, alo;
        load16(Qh + qrow, ahi);
        load16(Ql + qrow, alo);
        v8f s = t ? Sb : Sa;
        s = wmma_bf16(ahi.b, bhi.b, s);
        s = wmma_bf16(ahi.b, blo.b, s);
        s = wmma_bf16(alo.b, bhi.b, s);
        if (t) Sb = s; else Sa = s;
      }
    }

    // scale, causal mask, online softmax over l
    float Pa[8], Pb[8];
    float cmax = -1e30f;
#pragma unroll
    for (int r = 0; r < 8; ++r) {
      const int la = lb + r + 8 * hh;
      float va = Sa[r] * 0.125f;
      if (la > m_abs) va = -1e30f;
      Pa[r] = va; cmax = fmaxf(cmax, va);
      const int lbv = lb + 16 + r + 8 * hh;
      float vb = Sb[r] * 0.125f;
      if (lbv > m_abs) vb = -1e30f;
      Pb[r] = vb; cmax = fmaxf(cmax, vb);
    }
    cmax = fmaxf(cmax, __shfl_xor(cmax, 16, 32));
    const float new_max = fmaxf(run_max, cmax);
    const float corr = __expf(run_max - new_max);
    run_max = new_max;

    float lsum = 0.f;
#pragma unroll
    for (int r = 0; r < 8; ++r) {
      Pa[r] = __expf(Pa[r] - new_max); lsum += Pa[r];
      Pb[r] = __expf(Pb[r] - new_max); lsum += Pb[r];
    }
    lsum += __shfl_xor(lsum, 16, 32);
    run_sum = run_sum * corr + lsum;

#pragma unroll
    for (int dt = 0; dt < 4; ++dt)
#pragma unroll
      for (int r = 0; r < 8; ++r) O[dt][r] *= corr;

    // Build P as WMMA B operand (32 l x 16 m) via shfl_xor(16) half-exchange
    BF16x16 phi, plo;
#pragma unroll
    for (int r = 0; r < 8; ++r) {
      const float xa = __shfl_xor(Pa[r], 16, 32);
      const float xb = __shfl_xor(Pb[r], 16, 32);
      const float e_lo = hh ? xb    : Pa[r];
      const float e_hi = hh ? Pb[r] : xa;
      splitStore(e_lo, phi, plo, r);
      splitStore(e_hi, phi, plo, 8 + r);
    }

    // O += V(16d x 32l) * P(32l x 16m)
#pragma unroll
    for (int dt = 0; dt < 4; ++dt) {
      const size_t vrow = ((size_t)bh * DH_ + dt * 16 + lm) * L_ + lb + hh * 16;
      BF16x16 vhi, vlo;
      load16(Vh + vrow, vhi);
      load16(Vl + vrow, vlo);
      O[dt] = wmma_bf16(vhi.b, phi.b, O[dt]);
      O[dt] = wmma_bf16(vhi.b, plo.b, O[dt]);
      O[dt] = wmma_bf16(vlo.b, phi.b, O[dt]);
    }
  }

  const float inv = 1.f / run_sum;
  const int b = bh >> 4, h = bh & 15;
#pragma unroll
  for (int dt = 0; dt < 4; ++dt)
#pragma unroll
    for (int r = 0; r < 8; ++r) {
      const int d = dt * 16 + r + 8 * hh;
      const float v = O[dt][r] * inv;
      const unsigned short hv = f2bf(v);
      const unsigned short lv = f2bf(v - bf2f(hv));
      const size_t o = ((size_t)b * L_ + m_abs) * D_ + h * DH_ + d;   // catT (B,L,D)
      Ch[o] = hv; Cl[o] = lv;
    }
}

// ---------------------------------------------------------------------------
// Kernel 3: output projection  out[b] = Wo (D x D) @ cat[b] (D x L) + bo
// B operand from catT bf16 planes (contiguous feature dim).
// ---------------------------------------------------------------------------
__global__ void __launch_bounds__(256) out_proj_kernel(
    const unsigned short* __restrict__ Wop_hi, const unsigned short* __restrict__ Wop_lo,
    const unsigned short* __restrict__ Ch, const unsigned short* __restrict__ Cl,
    const float* __restrict__ bo, float* __restrict__ out)
{
  const int wid  = blockIdx.x * 8 + (threadIdx.x >> 5);
  const int lane = threadIdx.x & 31;
  const int lm = lane & 15, hh = lane >> 4;
  const int ntile = wid & 63;
  const int mb    = (wid >> 6) & 15;
  const int b     = wid >> 10;
  const int n_abs = ntile * 16 + lm;

  v8f acc[4];
#pragma unroll
  for (int mt = 0; mt < 4; ++mt)
#pragma unroll
    for (int r = 0; r < 8; ++r) acc[mt][r] = 0.f;

  for (int kt = 0; kt < 32; ++kt) {
    // B operand: catT row n_abs, 16 contiguous k
    const size_t brow = ((size_t)b * L_ + n_abs) * D_ + kt * 32 + hh * 16;
    BF16x16 bhi, blo;
    load16(Ch + brow, bhi);
    load16(Cl + brow, blo);
#pragma unroll
    for (int mt = 0; mt < 4; ++mt) {
      const size_t aoff = (((size_t)(mb * 4 + mt) * 32 + kt) * 32 + lane) * 16;
      BF16x16 ahi, alo;
      load16(Wop_hi + aoff, ahi);
      load16(Wop_lo + aoff, alo);
      acc[mt] = wmma_bf16(ahi.b, bhi.b, acc[mt]);
      acc[mt] = wmma_bf16(ahi.b, blo.b, acc[mt]);
      acc[mt] = wmma_bf16(alo.b, bhi.b, acc[mt]);
    }
  }

#pragma unroll
  for (int mt = 0; mt < 4; ++mt)
#pragma unroll
    for (int r = 0; r < 8; ++r) {
      const int m = mb * 64 + mt * 16 + r + 8 * hh;
      out[((size_t)b * D_ + m) * L_ + n_abs] = acc[mt][r] + bo[m];
    }
}

// ---------------------------------------------------------------------------
extern "C" void kernel_launch(void* const* d_in, const int* in_sizes, int n_in,
                              void* d_out, int out_size, void* d_ws, size_t ws_size,
                              hipStream_t stream) {
  const float* Q  = (const float*)d_in[0];
  const float* K  = (const float*)d_in[1];
  const float* V  = (const float*)d_in[2];
  const float* WQ = (const float*)d_in[3];
  const float* WK = (const float*)d_in[4];
  const float* WV = (const float*)d_in[5];
  const float* Wo = (const float*)d_in[6];
  const float* bo = (const float*)d_in[7];
  float* out = (float*)d_out;

  // workspace carve-up (all bf16/u16 elements)
  const size_t actE = (size_t)3 * B_ * D_ * L_;     // 12582912
  const size_t wE   = (size_t)3 * H_ * DH_ * D_;    // 3145728
  const size_t woE  = (size_t)D_ * D_;              // 1048576
  const size_t pE   = (size_t)B_ * H_ * L_ * DH_;   // 4194304

  unsigned short* p = (unsigned short*)d_ws;
  unsigned short* Xp_hi  = p;  p += actE;
  unsigned short* Xp_lo  = p;  p += actE;
  unsigned short* Wp_hi  = p;  p += wE;
  unsigned short* Wp_lo  = p;  p += wE;
  unsigned short* Wop_hi = p;  p += woE;
  unsigned short* Wop_lo = p;  p += woE;
  unsigned short* Qh     = p;  p += pE;
  unsigned short* Ql     = p;  p += pE;
  unsigned short* Kh     = p;  p += pE;
  unsigned short* Kl     = p;  p += pE;
  unsigned short* Vh     = p;  p += pE;
  unsigned short* Vl     = p;  p += pE;
  unsigned short* Ch     = p;  p += pE;
  unsigned short* Cl     = p;  p += pE;

  pack_act_kernel<<<(int)(actE / 256), 256, 0, stream>>>(Q, K, V, Xp_hi, Xp_lo);
  pack_w_kernel  <<<(int)(wE   / 256), 256, 0, stream>>>(WQ, WK, WV, Wp_hi, Wp_lo);
  pack_wo_kernel <<<(int)(woE  / 256), 256, 0, stream>>>(Wo, Wop_hi, Wop_lo);

  // 12288 waves: 64 bh * 3 mats * 64 ntiles -> 1536 blocks of 8 waves
  proj_qkv_kernel<<<1536, 256, 0, stream>>>(Xp_hi, Xp_lo, Wp_hi, Wp_lo,
                                            Qh, Ql, Kh, Kl, Vh, Vl);
  // 4096 waves: 64 bh * 64 mtiles -> 512 blocks
  attn_kernel<<<512, 256, 0, stream>>>(Qh, Ql, Kh, Kl, Vh, Vl, Ch, Cl);
  // 4096 waves: 4 b * 16 mblocks * 64 ntiles -> 512 blocks
  out_proj_kernel<<<512, 256, 0, stream>>>(Wop_hi, Wop_lo, Ch, Cl, bo, out);
}
